// Conv3x3Norm_8263517077878
// MI455X (gfx1250) — matrix operand order repaired and checked by reading, not compiled, not run on hardware
//
#include <hip/hip_runtime.h>
#include <math.h>

typedef __attribute__((ext_vector_type(16))) _Float16 v16h;
typedef __attribute__((ext_vector_type(2)))  _Float16 v2h;
typedef __attribute__((ext_vector_type(8)))  float    v8f;
typedef __attribute__((ext_vector_type(4)))  float    v4f;

#define BATCH 4
#define CIN   64
#define COUT  64
#define HH    128
#define WW    128
#define HW    (HH*WW)
#define TAPS  9
#define SB_STRIDE 72   // 64 ch + 8 halves pad -> 144B row stride, conflict-free

// ---------------------------------------------------------------------------
// Kernel 1: DCNv2 via implicit GEMM on WMMA f16 (f32 accumulate).
// Software-pipelined: double-buffered LDS so tap t+1 sampling (VALU+gather)
// co-executes with tap t WMMA (XDL); one barrier per tap.
// block = 256 threads (8 wave32) handles one output row (b, h): N-tile = 128
// ---------------------------------------------------------------------------
__global__ __launch_bounds__(256, 2)
void dcn_wmma_kernel(const float* __restrict__ x,
                     const float* __restrict__ offset,
                     const float* __restrict__ mask,
                     const float* __restrict__ weight,
                     const float* __restrict__ bias,
                     float* __restrict__ out) {
  // A fragments, pre-swizzled to contiguous-k A layout (same k order as sB reads):
  //   [buf][kk][mt][lane][i] ; element i of lane L -> K=(L/16)*16+i
  __shared__ __attribute__((aligned(32))) _Float16 sA[2][2*4*32*16];   // 16 KB
  // B panel: sampled+modulated patch, [buf][pixel][channel] f16
  __shared__ __attribute__((aligned(16))) _Float16 sB[2][128*SB_STRIDE]; // 36 KB

  const int tid = threadIdx.x;
  const int b   = blockIdx.x >> 7;
  const int h   = blockIdx.x & 127;

  // --- role: weight staging (256 threads cover kk(2) x mt(4) x lane(32)) ---
  const int wL  = tid & 31;
  const int wmt = (tid >> 5) & 3;
  const int wkk = tid >> 7;
  const int wM  = wmt * 16 + (wL & 15);
  const int whi = wL >> 4;

  // --- role: sampling (thread = (pixel n, channel-half)) ---
  const int n     = tid & 127;
  const int chalf = tid >> 7;
  const float* xb = x + (size_t)(b * CIN + chalf * 32) * HW;

  // --- role: WMMA (wave wv owns mt = wv&3, N-tiles ntb..ntb+3) ---
  const int wv    = tid >> 5;
  const int lane  = tid & 31;
  const int mt    = wv & 3;
  const int ntb   = (wv >> 2) * 4;
  const int nl    = lane & 15;
  const int khalf = lane >> 4;

  v8f acc[4];
  #pragma unroll
  for (int j = 0; j < 4; ++j) acc[j] = (v8f){0.f,0.f,0.f,0.f,0.f,0.f,0.f,0.f};

  // stage tap t (weights + bilinear-sampled modulated patch) into buffer bufi
  auto stage = [&](int t, int bufi) {
    // ---- A fragments ----
    {
      _Float16* a = &sA[bufi][((wkk * 4 + wmt) * 32 + wL) * 16];
      #pragma unroll
      for (int i = 0; i < 16; ++i) {
        const int Klocal = (whi << 4) | i;
        const int c = wkk * 32 + Klocal;
        a[i] = (_Float16)weight[(size_t)(wM * CIN + c) * 9 + t];
      }
    }
    // ---- bilinear sample + modulate into sB ----
    {
      const int ki = t / 3, kj = t % 3;
      const float offy = offset[((size_t)(b * 18 + 2 * t    ) * HH + h) * WW + n];
      const float offx = offset[((size_t)(b * 18 + 2 * t + 1) * HH + h) * WW + n];
      const float mval = mask  [((size_t)(b * 9  + t        ) * HH + h) * WW + n];
      const float py = (float)(h - 1 + ki) + offy;
      const float px = (float)(n - 1 + kj) + offx;
      const float y0f = floorf(py), x0f = floorf(px);
      const int   y0  = (int)y0f,   x0  = (int)x0f;
      const float wy = py - y0f, wx = px - x0f;
      const bool vy0 = (unsigned)y0       < (unsigned)HH;
      const bool vy1 = (unsigned)(y0 + 1) < (unsigned)HH;
      const bool vx0 = (unsigned)x0       < (unsigned)WW;
      const bool vx1 = (unsigned)(x0 + 1) < (unsigned)WW;
      float w00 = (1.f - wy) * (1.f - wx) * mval; if (!(vy0 && vx0)) w00 = 0.f;
      float w01 = (1.f - wy) * wx         * mval; if (!(vy0 && vx1)) w01 = 0.f;
      float w10 = wy * (1.f - wx)         * mval; if (!(vy1 && vx0)) w10 = 0.f;
      float w11 = wy * wx                 * mval; if (!(vy1 && vx1)) w11 = 0.f;
      const int y0c = min(max(y0, 0), HH - 1), y1c = min(max(y0 + 1, 0), HH - 1);
      const int x0c = min(max(x0, 0), WW - 1), x1c = min(max(x0 + 1, 0), WW - 1);
      const int i00 = y0c * WW + x0c, i01 = y0c * WW + x1c;
      const int i10 = y1c * WW + x0c, i11 = y1c * WW + x1c;
      _Float16* srow = &sB[bufi][n * SB_STRIDE + chalf * 32];
      #pragma unroll 4
      for (int c2 = 0; c2 < 16; ++c2) {
        const float* p0 = xb + (size_t)(2 * c2) * HW;
        const float* p1 = p0 + HW;
        const float va = w00*p0[i00] + w01*p0[i01] + w10*p0[i10] + w11*p0[i11];
        const float vb = w00*p1[i00] + w01*p1[i01] + w10*p1[i10] + w11*p1[i11];
        v2h hv; hv.x = (_Float16)va; hv.y = (_Float16)vb;
        *(v2h*)(srow + 2 * c2) = hv;
      }
    }
  };

  stage(0, 0);
  __syncthreads();

  for (int t = 0; t < TAPS; ++t) {
    const int cur = t & 1;
    // ---- WMMA on tap t: D += A(16x32) x B(32x16) ----
    #pragma unroll
    for (int kk = 0; kk < 2; ++kk) {
      const v16h a = *(const v16h*)&sA[cur][((kk * 4 + mt) * 32 + lane) * 16];
      const int cstart = kk * 32 + khalf * 16;
      #pragma unroll
      for (int j = 0; j < 4; ++j) {
        const int pix = (ntb + j) * 16 + nl;
        const v16h bf = *(const v16h*)&sB[cur][pix * SB_STRIDE + cstart];
        acc[j] = __builtin_amdgcn_wmma_f32_16x16x32_f16(
            false, a, false, bf, (short)0, acc[j], false, false);
      }
    }
    // ---- stage tap t+1 into the other buffer (overlaps with WMMA above) ----
    if (t + 1 < TAPS) stage(t + 1, (t + 1) & 1);
    __syncthreads();
  }

  // ---- epilogue: bias + store (C/D layout: M = r + (lane>=16)*8) ----
  #pragma unroll
  for (int r = 0; r < 8; ++r) {
    const int o  = mt * 16 + r + khalf * 8;
    const float bv = bias[o];
    #pragma unroll
    for (int j = 0; j < 4; ++j) {
      const int pix = (ntb + j) * 16 + nl;
      out[((size_t)(b * COUT + o) * HH + h) * WW + pix] = acc[j][r] + bv;
    }
  }
}

// ---------------------------------------------------------------------------
// Kernel 2: GroupNorm statistics. Groups of 4 channels are contiguous 256KB
// spans in NCHW; one block per (b, group) -> mean & rsqrt(var+eps) in d_ws.
// ---------------------------------------------------------------------------
__global__ __launch_bounds__(256)
void gn_stats_kernel(const float* __restrict__ y, float* __restrict__ stats) {
  __shared__ float s1[256];
  __shared__ float s2[256];
  const int tid = threadIdx.x;
  const v4f* p = (const v4f*)(y + (size_t)blockIdx.x * 65536);
  float s = 0.f, q = 0.f;
  for (int i = tid; i < 16384; i += 256) {
    const v4f v = p[i];
    s += v.x + v.y + v.z + v.w;
    q += v.x * v.x + v.y * v.y + v.z * v.z + v.w * v.w;
  }
  s1[tid] = s; s2[tid] = q;
  __syncthreads();
  for (int o = 128; o > 0; o >>= 1) {
    if (tid < o) { s1[tid] += s1[tid + o]; s2[tid] += s2[tid + o]; }
    __syncthreads();
  }
  if (tid == 0) {
    const float inv = 1.f / 65536.f;
    const float m   = s1[0] * inv;
    const float var = s2[0] * inv - m * m;
    stats[2 * blockIdx.x]     = m;
    stats[2 * blockIdx.x + 1] = rsqrtf(var + 1e-5f);
  }
}

// ---------------------------------------------------------------------------
// Kernel 3: apply GroupNorm affine in place (float4 vectorized).
// ---------------------------------------------------------------------------
__global__ __launch_bounds__(256)
void gn_apply_kernel(float* __restrict__ y, const float* __restrict__ stats,
                     const float* __restrict__ gamma, const float* __restrict__ beta) {
  const int idx4 = blockIdx.x * 256 + threadIdx.x;
  if (idx4 >= (BATCH * COUT * HW) / 4) return;
  const int gi = idx4 >> 14;         // (b*16 + group): 16384 float4 per group
  const int c  = (idx4 >> 12) & 63;  // channel (4096 float4 per channel)
  const float m  = stats[2 * gi];
  const float rs = stats[2 * gi + 1];
  const float g  = gamma[c];
  const float be = beta[c];
  v4f v = ((v4f*)y)[idx4];
  v.x = (v.x - m) * rs * g + be;
  v.y = (v.y - m) * rs * g + be;
  v.z = (v.z - m) * rs * g + be;
  v.w = (v.w - m) * rs * g + be;
  ((v4f*)y)[idx4] = v;
}

extern "C" void kernel_launch(void* const* d_in, const int* in_sizes, int n_in,
                              void* d_out, int out_size, void* d_ws, size_t ws_size,
                              hipStream_t stream) {
  (void)in_sizes; (void)n_in; (void)out_size; (void)ws_size;
  const float* x      = (const float*)d_in[0];
  const float* offset = (const float*)d_in[1];
  const float* mask   = (const float*)d_in[2];
  const float* weight = (const float*)d_in[3];
  const float* bias   = (const float*)d_in[4];
  const float* gamma  = (const float*)d_in[5];
  const float* beta   = (const float*)d_in[6];
  float* out   = (float*)d_out;
  float* stats = (float*)d_ws;   // 128 floats: (mean, rstd) x 64 groups

  dcn_wmma_kernel<<<BATCH * HH, 256, 0, stream>>>(x, offset, mask, weight, bias, out);
  gn_stats_kernel<<<BATCH * 16, 256, 0, stream>>>(out, stats);
  gn_apply_kernel<<<(BATCH * COUT * HW) / 4 / 256, 256, 0, stream>>>(out, stats, gamma, beta);
}
